// TransformerBlockQuantum_65481071398402
// MI455X (gfx1250) — matrix operand order repaired and checked by reading, not compiled, not run on hardware
//
#include <hip/hip_runtime.h>
#include <hip/hip_bf16.h>
#include <math.h>

// ---- problem constants (match reference) ----
#define BB   4
#define SS   1024
#define EE   256
#define HH   32
#define DKK  8
#define NQQ  8
#define FFF  1024
#define NT   (BB*SS)          // 4096 tokens
#define EPSF 1e-5f

// GEMM blocking: block = 8 waves as 2x4 grid, each wave owns 2x2 16x16 tiles
// => block tile = 64 (M) x 128 (N). K staged in chunks of KC=4 ksteps (128 K).
#define KC        4
#define A_CH_HW   (4*KC*512)          // 8192 halfwords = 16KB  (4 mtiles)
#define B_CH_HW   (KC*8*512)          // 16384 halfwords = 32KB (8 ntiles)
#define CH_HW     (A_CH_HW + B_CH_HW) // 24576 halfwords = 48KB
#define CH_BYTES  (CH_HW*2)

typedef __bf16 v16bf __attribute__((ext_vector_type(16)));
typedef float  v8f   __attribute__((ext_vector_type(8)));
typedef unsigned int u32x4 __attribute__((ext_vector_type(4)));
typedef int i32x8 __attribute__((ext_vector_type(8)));
typedef int i32x4 __attribute__((ext_vector_type(4)));

#if defined(__has_builtin)
#if __has_builtin(__builtin_amdgcn_tensor_load_to_lds) && __has_builtin(__builtin_amdgcn_s_wait_tensorcnt)
#define HAS_TDM 1
#endif
#endif
#ifndef HAS_TDM
#define HAS_TDM 0
#endif

// float -> bf16 (round to nearest even)
__device__ __forceinline__ unsigned short f2bf(float f) {
  unsigned u = __float_as_uint(f);
  u += 0x7FFFu + ((u >> 16) & 1u);
  return (unsigned short)(u >> 16);
}

// A-fragment (16x32 bf16) k-local index for (lane, halfword e)  [ISA 7.12.2]
__device__ __forceinline__ int a_klocal(int lane, int e) {
  int half = lane >> 4;
  int r = e >> 1;
  int p = e & 1;
  int kb = ((r & 4) ? 16 : 0) + half * 8 + (r & 3) * 2;
  return kb + p;
}
// inverse: k_local in [0,32) -> (lane_half, e)
__device__ __forceinline__ void a_kinv(int kl, int& half, int& e) {
  half = (kl >> 3) & 1;
  int r = ((kl >= 16) ? 4 : 0) + ((kl & 7) >> 1);
  e = 2 * r + (kl & 1);
}
// B-fragment (32x16 bf16) k-local index for (lane, halfword j)
__device__ __forceinline__ int b_klocal(int lane, int j) {
  return ((j >> 3) << 4) + ((lane >> 4) << 3) + (j & 7);
}

#if HAS_TDM
// Issue one 2-D TDM tile load: `tile1` rows of `tile0` dwords, global row
// stride `stride0_dw` dwords, into contiguous LDS at byte offset `lds_addr`.
// D# bitfields per CDNA5 ISA (data_size=4B).
__device__ __forceinline__ void tdm_load_2d(unsigned lds_addr, const void* gaddr,
                                            unsigned tile0, unsigned tile1,
                                            unsigned long long stride0_dw) {
  unsigned long long ga = (unsigned long long)gaddr;
  u32x4 g0;
  g0.x = 1u;                                  // count=1, user descriptor
  g0.y = lds_addr;                            // lds_addr (bytes)
  g0.z = (unsigned)ga;                        // global_addr[31:0]
  g0.w = (unsigned)(ga >> 32) | 0x80000000u;  // global_addr[56:32] | type=2<<30
  i32x8 g1;
  g1[0] = 0x20000;                                            // data_size=2 (4B)
  g1[1] = (int)((tile0 & 0xFFFFu) << 16);                     // tensor_dim0[15:0]
  g1[2] = (int)((tile0 >> 16) | ((tile1 & 0xFFFFu) << 16));   // tdim0 hi | tdim1 lo
  g1[3] = (int)((tile1 >> 16) | (tile0 << 16));               // tdim1 hi | tile_dim0
  g1[4] = (int)(tile1 & 0xFFFFu);                             // tile_dim1 (tile_dim2=0)
  g1[5] = (int)(stride0_dw & 0xFFFFFFFFull);                  // tensor_dim0_stride lo
  g1[6] = (int)((stride0_dw >> 32) & 0xFFFFull);              // stride hi (dim1_stride=0)
  g1[7] = 0;
  i32x4 z = {0, 0, 0, 0};
#if __clang_major__ >= 23
  i32x8 z8 = {0, 0, 0, 0, 0, 0, 0, 0};
  __builtin_amdgcn_tensor_load_to_lds(g0, g1, z, z, z8, 0);
#else
  __builtin_amdgcn_tensor_load_to_lds(g0, g1, z, z, 0);
#endif
}
#endif

// ---------------------------------------------------------------------------
// 1) pack x -> A-fragments of q_out = cos(x + theta_attn[k%8])  (bf16)
// ---------------------------------------------------------------------------
__global__ __launch_bounds__(256) void k_pack_qout(const float* __restrict__ x,
                                                   const float* __restrict__ theta,
                                                   unsigned short* __restrict__ ap) {
  int t = blockIdx.x * 256 + threadIdx.x;           // over NT*EE
  int e     = t & 15;
  int lane  = (t >> 4) & 31;
  int kstep = (t >> 9) & 7;                         // EE/32 = 8
  int mtile = t >> 12;
  int m = mtile * 16 + (lane & 15);
  int k = kstep * 32 + a_klocal(lane, e);
  float v = __cosf(x[(size_t)m * EE + k] + theta[k & 7]);
  ap[t] = f2bf(v);
}

// ---------------------------------------------------------------------------
// 2) pack weight w[N,K] (row-major fp32) -> B-fragments bf16
//    Bpacked layout: [Ksteps][Ntiles][lane=32][j=16]; B(k,n) = w[n,k]
// ---------------------------------------------------------------------------
__global__ __launch_bounds__(256) void k_pack_w(const float* __restrict__ w,
                                                unsigned short* __restrict__ bp,
                                                int N, int K) {
  int t = blockIdx.x * 256 + threadIdx.x;           // over K*N
  int j    = t & 15;
  int lane = (t >> 4) & 31;
  int rest = t >> 9;
  int ntiles = N >> 4;
  int ntile = rest % ntiles;
  int kstep = rest / ntiles;
  int n = ntile * 16 + (lane & 15);
  int k = kstep * 32 + b_klocal(lane, j);
  bp[t] = f2bf(w[(size_t)n * K + k]);
}

// ---------------------------------------------------------------------------
// 3) fold out-proj into combine: wcmb[n,k] = sum_j w_comb[n,j]*w_out[j,k]
// ---------------------------------------------------------------------------
__global__ __launch_bounds__(256) void k_fold_w(const float* __restrict__ w_comb,
                                                const float* __restrict__ w_out,
                                                float* __restrict__ wcmb) {
  int idx = blockIdx.x * 256 + threadIdx.x;         // over 256*256
  int k = idx & 255, n = idx >> 8;
  float acc = 0.f;
  #pragma unroll 8
  for (int j = 0; j < EE; ++j)
    acc = fmaf(w_comb[n * EE + j], w_out[j * EE + k], acc);
  wcmb[idx] = acc;
}
__global__ void k_fold_b(const float* __restrict__ w_comb,
                         const float* __restrict__ b_out,
                         const float* __restrict__ b_comb,
                         float* __restrict__ bc2) {
  int n = threadIdx.x;
  float acc = b_comb[n];
  for (int j = 0; j < EE; ++j) acc = fmaf(w_comb[n * EE + j], b_out[j], acc);
  bc2[n] = acc;
}

// ---------------------------------------------------------------------------
// 4) WMMA GEMM, TDM-staged: block = 8 waves (2x4), wave = 2x2 tiles.
//    A/B panels DMA'd to LDS per K-chunk (double buffered), fragments read
//    with ds_load_b128, 4 wmma per kstep per wave.
// ---------------------------------------------------------------------------
__global__ __launch_bounds__(256) void k_gemm_tdm(const unsigned short* __restrict__ Ap,
                                                  const unsigned short* __restrict__ Bp,
                                                  const float* __restrict__ bias,
                                                  float* __restrict__ out,
                                                  int N, int K) {
  __shared__ unsigned short ldsm[2 * CH_HW];        // 96 KB (two chunks)
  int wave = threadIdx.x >> 5;
  int lane = threadIdx.x & 31;
  int wm = wave >> 2, wn = wave & 3;
  int ntiles = N >> 4;
  int ksteps = K >> 5;
  int nblocks = ntiles >> 3;
  int mt0 = (blockIdx.x / nblocks) * 4;
  int nt0 = (blockIdx.x % nblocks) * 8;
  int nchunks = ksteps / KC;

  v8f c00 = {0.f,0.f,0.f,0.f,0.f,0.f,0.f,0.f};
  v8f c01 = c00, c10 = c00, c11 = c00;

#if HAS_TDM
  unsigned lds_base = (unsigned)(unsigned long long)(void*)ldsm;
  if (wave == 0) {
    // prologue: chunk 0 -> buffer 0
    const void* ga = Ap + (((size_t)mt0 * ksteps) * 512);
    const void* gb = Bp + ((size_t)nt0 * 512);
    tdm_load_2d(lds_base,               ga, KC * 256, 4,  (unsigned long long)ksteps * 256);
    tdm_load_2d(lds_base + A_CH_HW * 2, gb, 2048,     KC, (unsigned long long)ntiles * 256);
  }
#endif

  for (int c = 0; c < nchunks; ++c) {
#if HAS_TDM
    int buf = c & 1;
    if (wave == 0) {
      if (c + 1 < nchunks) {
        int kc0 = (c + 1) * KC;
        unsigned lb = lds_base + (unsigned)((c + 1) & 1) * CH_BYTES;
        const void* ga = Ap + (((size_t)mt0 * ksteps + kc0) * 512);
        const void* gb = Bp + (((size_t)kc0 * ntiles + nt0) * 512);
        tdm_load_2d(lb,               ga, KC * 256, 4,  (unsigned long long)ksteps * 256);
        tdm_load_2d(lb + A_CH_HW * 2, gb, 2048,     KC, (unsigned long long)ntiles * 256);
        __builtin_amdgcn_s_wait_tensorcnt(2);   // chunk c complete (in-order), c+1 in flight
      } else {
        __builtin_amdgcn_s_wait_tensorcnt(0);
      }
    }
    __syncthreads();                            // chunk c visible to all waves
    const unsigned short* lb = &ldsm[buf * CH_HW];
#else
    __syncthreads();
    {   // cooperative fallback copy, single buffer
      int kc0 = c * KC;
      unsigned* la = (unsigned*)&ldsm[0];
      const unsigned* gaA = (const unsigned*)Ap;
      const unsigned* gaB = (const unsigned*)Bp;
      for (int i = threadIdx.x; i < 4 * KC * 256; i += 256) {
        int row = i / (KC * 256), col = i % (KC * 256);
        la[i] = gaA[((size_t)(mt0 + row) * ksteps + kc0) * 256 + col];
      }
      unsigned* lbp = (unsigned*)&ldsm[A_CH_HW];
      for (int i = threadIdx.x; i < KC * 8 * 256; i += 256) {
        int row = i / (8 * 256), col = i % (8 * 256);
        lbp[i] = gaB[((size_t)(kc0 + row) * ntiles + nt0) * 256 + col];
      }
    }
    __syncthreads();
    const unsigned short* lb = &ldsm[0];
#endif

    #pragma unroll
    for (int ks = 0; ks < KC; ++ks) {
      int mi0 = 2 * wm, ni0 = 2 * wn;
      v16bf a0 = *reinterpret_cast<const v16bf*>(lb + (((mi0    ) * KC + ks) * 32 + lane) * 16);
      v16bf a1 = *reinterpret_cast<const v16bf*>(lb + (((mi0 + 1) * KC + ks) * 32 + lane) * 16);
      v16bf b0 = *reinterpret_cast<const v16bf*>(lb + A_CH_HW + ((ks * 8 + ni0    ) * 32 + lane) * 16);
      v16bf b1 = *reinterpret_cast<const v16bf*>(lb + A_CH_HW + ((ks * 8 + ni0 + 1) * 32 + lane) * 16);
      c00 = __builtin_amdgcn_wmma_f32_16x16x32_bf16(false, a0, false, b0, (short)0, c00, false, false);
      c01 = __builtin_amdgcn_wmma_f32_16x16x32_bf16(false, a0, false, b1, (short)0, c01, false, false);
      c10 = __builtin_amdgcn_wmma_f32_16x16x32_bf16(false, a1, false, b0, (short)0, c10, false, false);
      c11 = __builtin_amdgcn_wmma_f32_16x16x32_bf16(false, a1, false, b1, (short)0, c11, false, false);
    }
    __syncthreads();                            // done reading buf before TDM reuses it
  }

  // epilogue: C/D layout n = lane%16, m = r + 8*(lane/16)
  int nloc = lane & 15, mh = lane >> 4;
  v8f* accs[2][2] = {{&c00, &c01}, {&c10, &c11}};
  #pragma unroll
  for (int mi = 0; mi < 2; ++mi) {
    #pragma unroll
    for (int ni = 0; ni < 2; ++ni) {
      int mtile = mt0 + 2 * wm + mi;
      int ntile = nt0 + 2 * wn + ni;
      int n = ntile * 16 + nloc;
      float bv = bias[n];
      v8f cc = *accs[mi][ni];
      #pragma unroll
      for (int r = 0; r < 8; ++r) {
        int m = mtile * 16 + r + mh * 8;
        out[(size_t)m * N + n] = cc[r] + bv;
      }
    }
  }
}

// ---------------------------------------------------------------------------
// 5) attention core (dk=8): one thread per (b,h,query). Two-pass softmax.
//    Writes ctx directly in A-fragment packed bf16 order for the next GEMM.
// ---------------------------------------------------------------------------
__global__ __launch_bounds__(256) void k_attention(const float* __restrict__ qkv,
                                                   unsigned short* __restrict__ ctxA) {
  int idx = blockIdx.x * 256 + threadIdx.x;         // over B*H*S = 131072
  int b = idx >> 15;
  int h = (idx >> 10) & 31;
  int s = idx & 1023;
  int m = b * SS + s;

  const float* qp = qkv + (size_t)m * 768 + h * DKK;
  float q[DKK];
  #pragma unroll
  for (int d = 0; d < DKK; ++d) q[d] = qp[d];
  const float scale = 0.35355339059327373f;         // 1/sqrt(8)

  const float* kbase = qkv + (size_t)(b * SS) * 768 + 256 + h * DKK;
  const float* vbase = qkv + (size_t)(b * SS) * 768 + 512 + h * DKK;

  float mx = -3.4e38f;
  for (int j = 0; j < SS; ++j) {
    const float* kp = kbase + (size_t)j * 768;
    float dot = 0.f;
    #pragma unroll
    for (int d = 0; d < DKK; ++d) dot = fmaf(q[d], kp[d], dot);
    mx = fmaxf(mx, dot * scale);
  }
  float se = 0.f, vacc[DKK];
  #pragma unroll
  for (int d = 0; d < DKK; ++d) vacc[d] = 0.f;
  for (int j = 0; j < SS; ++j) {
    const float* kp = kbase + (size_t)j * 768;
    const float* vp = vbase + (size_t)j * 768;
    float dot = 0.f;
    #pragma unroll
    for (int d = 0; d < DKK; ++d) dot = fmaf(q[d], kp[d], dot);
    float e = __expf(dot * scale - mx);
    se += e;
    #pragma unroll
    for (int d = 0; d < DKK; ++d) vacc[d] = fmaf(e, vp[d], vacc[d]);
  }
  float inv = 1.f / se;

  int mtile = m >> 4;
  #pragma unroll
  for (int d = 0; d < DKK; ++d) {
    int k = h * DKK + d;
    int kstep = k >> 5, kl = k & 31, half, e;
    a_kinv(kl, half, e);
    int lane = half * 16 + (m & 15);
    ctxA[(((size_t)mtile * 8 + kstep) * 32 + lane) * 16 + e] = f2bf(vacc[d] * inv);
  }
}

// ---------------------------------------------------------------------------
// 6) layernorm over E=256 with residual: out = LN(a + r)*g + be
// ---------------------------------------------------------------------------
__global__ __launch_bounds__(256) void k_ln(const float* __restrict__ a,
                                            const float* __restrict__ r,
                                            const float* __restrict__ g,
                                            const float* __restrict__ be,
                                            float* __restrict__ out) {
  __shared__ float sred[256];
  int row = blockIdx.x, tid = threadIdx.x;
  float v = a[(size_t)row * EE + tid] + r[(size_t)row * EE + tid];
  sred[tid] = v; __syncthreads();
  for (int s = 128; s > 0; s >>= 1) { if (tid < s) sred[tid] += sred[tid + s]; __syncthreads(); }
  float mu = sred[0] * (1.f / EE); __syncthreads();
  float d = v - mu;
  sred[tid] = d * d; __syncthreads();
  for (int s = 128; s > 0; s >>= 1) { if (tid < s) sred[tid] += sred[tid + s]; __syncthreads(); }
  float var = sred[0] * (1.f / EE);
  out[(size_t)row * EE + tid] = d * rsqrtf(var + EPSF) * g[tid] + be[tid];
}

// ---------------------------------------------------------------------------
// 7) qf = cos(theta_ffn) * cos(h @ w_ip.T + b_ip)   [NT, 8]
// ---------------------------------------------------------------------------
__global__ __launch_bounds__(256) void k_proj_qf(const float* __restrict__ h,
                                                 const float* __restrict__ w_ip,
                                                 const float* __restrict__ b_ip,
                                                 const float* __restrict__ th,
                                                 float* __restrict__ qf) {
  int idx = blockIdx.x * 256 + threadIdx.x;         // over NT*8
  int nq = idx & 7, m = idx >> 3;
  float acc = b_ip[nq];
  const float* hr = h + (size_t)m * EE;
  const float* wr = w_ip + (size_t)nq * EE;
  #pragma unroll 8
  for (int k = 0; k < EE; ++k) acc = fmaf(hr[k], wr[k], acc);
  qf[idx] = __cosf(th[nq]) * __cosf(acc);
}

// ---------------------------------------------------------------------------
// 8) relu1 = relu(qf @ w1.T + b1), written directly in A-fragment bf16 order
// ---------------------------------------------------------------------------
__global__ __launch_bounds__(256) void k_ffn1(const float* __restrict__ qf,
                                              const float* __restrict__ w1,
                                              const float* __restrict__ b1,
                                              unsigned short* __restrict__ ap) {
  int t = blockIdx.x * 256 + threadIdx.x;           // over NT*FF
  int e    = t & 15;
  int lane = (t >> 4) & 31;
  int rest = t >> 9;
  int kstep = rest & 31;                            // FF/32 = 32
  int mtile = rest >> 5;
  int m = mtile * 16 + (lane & 15);
  int ff = kstep * 32 + a_klocal(lane, e);
  float acc = b1[ff];
  const float* qr = qf + (size_t)m * NQQ;
  const float* wr = w1 + (size_t)ff * NQQ;
  #pragma unroll
  for (int q = 0; q < NQQ; ++q) acc = fmaf(qr[q], wr[q], acc);
  ap[t] = f2bf(fmaxf(acc, 0.f));
}

// ---------------------------------------------------------------------------
extern "C" void kernel_launch(void* const* d_in, const int* in_sizes, int n_in,
                              void* d_out, int out_size, void* d_ws, size_t ws_size,
                              hipStream_t stream) {
  const float* x        = (const float*)d_in[0];
  const float* th_attn  = (const float*)d_in[1];
  const float* w_in     = (const float*)d_in[2];
  const float* b_in     = (const float*)d_in[3];
  const float* w_out    = (const float*)d_in[4];
  const float* b_out    = (const float*)d_in[5];
  const float* w_comb   = (const float*)d_in[6];
  const float* b_comb   = (const float*)d_in[7];
  const float* g1       = (const float*)d_in[8];
  const float* be1      = (const float*)d_in[9];
  const float* g2       = (const float*)d_in[10];
  const float* be2      = (const float*)d_in[11];
  const float* w_ip     = (const float*)d_in[12];
  const float* b_ip     = (const float*)d_in[13];
  const float* th_ffn   = (const float*)d_in[14];
  const float* w1       = (const float*)d_in[15];
  const float* b1       = (const float*)d_in[16];
  const float* w2       = (const float*)d_in[17];
  const float* b2       = (const float*)d_in[18];
  float* out            = (float*)d_out;

  // ---- workspace carve-up (256B aligned) ----
  char* ws = (char*)d_ws;
  size_t off = 0;
  auto alloc = [&](size_t bytes) { char* p = ws + off; off = (off + bytes + 255) & ~(size_t)255; return p; };
  unsigned short* qoutA  = (unsigned short*)alloc((size_t)NT * EE * 2);       // 2 MB
  unsigned short* winB   = (unsigned short*)alloc((size_t)EE * 768 * 2);      // 384 KB
  float*          wcmb   = (float*)alloc((size_t)EE * EE * 4);                // 256 KB
  float*          bc2    = (float*)alloc((size_t)EE * 4);
  unsigned short* wcmbB  = (unsigned short*)alloc((size_t)EE * EE * 2);
  unsigned short* w2B    = (unsigned short*)alloc((size_t)FFF * EE * 2);
  float*          qkv    = (float*)alloc((size_t)NT * 768 * 4);               // 12 MB
  unsigned short* ctxA   = (unsigned short*)alloc((size_t)NT * EE * 2);
  float*          attn_o = (float*)alloc((size_t)NT * EE * 4);
  float*          hbuf   = (float*)alloc((size_t)NT * EE * 4);
  float*          qf     = (float*)alloc((size_t)NT * NQQ * 4);
  unsigned short* relu1A = (unsigned short*)alloc((size_t)NT * FFF * 2);      // 8 MB
  float*          ffn_o  = (float*)alloc((size_t)NT * EE * 4);
  (void)ws_size; (void)in_sizes; (void)n_in; (void)out_size;

  // ---- operand prep (cheap, L2-resident) ----
  k_pack_qout<<<(NT * EE) / 256, 256, 0, stream>>>(x, th_attn, qoutA);
  k_pack_w<<<(EE * 768) / 256, 256, 0, stream>>>(w_in, winB, 768, EE);
  k_fold_w<<<(EE * EE) / 256, 256, 0, stream>>>(w_comb, w_out, wcmb);
  k_fold_b<<<1, 256, 0, stream>>>(w_comb, b_out, b_comb, bc2);
  k_pack_w<<<(EE * EE) / 256, 256, 0, stream>>>(wcmb, wcmbB, EE, EE);
  k_pack_w<<<(FFF * EE) / 256, 256, 0, stream>>>(w2, w2B, EE, FFF);

  // ---- attention path ----
  // qkv = cos(x+theta) @ w_in.T + b_in   (WMMA+TDM, M=4096 N=768 K=256)
  k_gemm_tdm<<<(256 / 4) * (48 / 8), 256, 0, stream>>>(qoutA, winB, b_in, qkv, 768, EE);
  k_attention<<<(BB * HH * SS) / 256, 256, 0, stream>>>(qkv, ctxA);
  // attn_out = ctx @ (w_comb.w_out).T + bc2  (WMMA+TDM, M=4096 N=256 K=256)
  k_gemm_tdm<<<(256 / 4) * (16 / 8), 256, 0, stream>>>(ctxA, wcmbB, bc2, attn_o, EE, EE);
  k_ln<<<NT, 256, 0, stream>>>(x, attn_o, g1, be1, hbuf);

  // ---- quantum FFN path ----
  k_proj_qf<<<(NT * NQQ) / 256, 256, 0, stream>>>(hbuf, w_ip, b_ip, th_ffn, qf);
  k_ffn1<<<(NT * FFF) / 256, 256, 0, stream>>>(qf, w1, b1, relu1A);
  // ffn_out = relu1 @ w2.T + b2          (WMMA+TDM, M=4096 N=256 K=1024)
  k_gemm_tdm<<<(256 / 4) * (16 / 8), 256, 0, stream>>>(relu1A, w2B, b2, ffn_o, EE, FFF);
  k_ln<<<NT, 256, 0, stream>>>(hbuf, ffn_o, g2, be2, out);
}
